// FASTOPIC_39754217291888
// MI455X (gfx1250) — matrix operand order repaired and checked by reading, not compiled, not run on hardware
//
#include <hip/hip_runtime.h>
#include <hip/hip_fp16.h>
#include <math.h>

#define N_DOCS   4096
#define N_VOCAB  50000
#define N_TOPIC  50
#define N_DIM    384
#define KPAD     64
#define ITERS    100
#define SEPS     1e-16f
#define REPS     1e-12f
#define NSHARD   32        // loss accumulator shards (slots 8..39 of loss block)

typedef __attribute__((ext_vector_type(16))) _Float16 v16h;
typedef __attribute__((ext_vector_type(8)))  _Float16 v8h;
typedef __attribute__((ext_vector_type(8)))  float    v8f;

// ---------------- utility device functions ----------------

__device__ __forceinline__ float wave_sum(float v) {
#pragma unroll
    for (int off = 16; off > 0; off >>= 1) v += __shfl_down(v, off, 32);
    return v;
}

__device__ __forceinline__ float devload_agent(const float* p) {
    return __hip_atomic_load(p, __ATOMIC_RELAXED, __HIP_MEMORY_SCOPE_AGENT);
}

// software grid barrier (all blocks must be resident; tiny grids used)
__device__ __forceinline__ void grid_barrier(unsigned* cnt, unsigned* gen, unsigned nblk) {
    __syncthreads();
    if (threadIdx.x == 0) {
        __threadfence();
        volatile unsigned* vgen = (volatile unsigned*)gen;
        unsigned g = *vgen;
        unsigned arrived = atomicAdd(cnt, 1u);
        if (arrived == nblk - 1u) {
            *(volatile unsigned*)cnt = 0u;
            __threadfence();
            atomicAdd(gen, 1u);
        } else {
            while (*vgen == g) { __builtin_amdgcn_s_sleep(1); }
        }
        __threadfence();
    }
    __syncthreads();
}

// ---------------- init: zero accumulators / theta / beta / loss / barriers ----------------

__global__ void fastopic_zero(unsigned* __restrict__ p, size_t nwords) {
    size_t i = (size_t)blockIdx.x * blockDim.x + threadIdx.x;
    size_t stride = (size_t)gridDim.x * blockDim.x;
    for (; i < nwords; i += stride) p[i] = 0u;
}

// ---------------- softmaxes of word / topic weights ----------------

__global__ void __launch_bounds__(1024)
fastopic_softmax(const float* __restrict__ wwt, const float* __restrict__ twt,
                 float* __restrict__ bw, float* __restrict__ btop) {
    __shared__ float red[1024];
    const int tid = threadIdx.x;
    float m = -3.4e38f;
    for (int j = tid; j < N_VOCAB; j += 1024) m = fmaxf(m, wwt[j]);
    red[tid] = m; __syncthreads();
    for (int s = 512; s > 0; s >>= 1) { if (tid < s) red[tid] = fmaxf(red[tid], red[tid + s]); __syncthreads(); }
    const float mx = red[0]; __syncthreads();
    float ssum = 0.f;
    for (int j = tid; j < N_VOCAB; j += 1024) ssum += __expf(wwt[j] - mx);
    red[tid] = ssum; __syncthreads();
    for (int s = 512; s > 0; s >>= 1) { if (tid < s) red[tid] += red[tid + s]; __syncthreads(); }
    const float inv = 1.0f / red[0]; __syncthreads();
    for (int j = tid; j < N_VOCAB; j += 1024) bw[j] = __expf(wwt[j] - mx) * inv;
    if (tid == 0) {
        float tm = -3.4e38f;
        for (int t = 0; t < N_TOPIC; ++t) tm = fmaxf(tm, twt[t]);
        float ts = 0.f;
        for (int t = 0; t < N_TOPIC; ++t) ts += __expf(twt[t] - tm);
        for (int t = 0; t < N_TOPIC; ++t) btop[t] = __expf(twt[t] - tm) / ts;
    }
}

// ---------------- K/M for doc-topic (stored transposed [t][i] for coalesced Sinkhorn) ----------------

__global__ void __launch_bounds__(256)
fastopic_kdt(const float* __restrict__ doc, const float* __restrict__ topic,
             float* __restrict__ KT, float* __restrict__ MT) {
    __shared__ float tsh[N_TOPIC * N_DIM];
    __shared__ float yy[N_TOPIC];
    for (int idx = threadIdx.x; idx < N_TOPIC * N_DIM; idx += 256) tsh[idx] = topic[idx];
    __syncthreads();
    if (threadIdx.x < N_TOPIC) {
        float s = 0.f;
        for (int k = 0; k < N_DIM; ++k) { float t = tsh[threadIdx.x * N_DIM + k]; s += t * t; }
        yy[threadIdx.x] = s;
    }
    __syncthreads();
    const int i = blockIdx.x * 256 + threadIdx.x;      // < 4096
    const float* x = doc + (size_t)i * N_DIM;
    float acc[N_TOPIC];
#pragma unroll
    for (int t = 0; t < N_TOPIC; ++t) acc[t] = 0.f;
    float xx = 0.f;
    for (int kc = 0; kc < N_DIM; kc += 8) {
        float xv[8];
#pragma unroll
        for (int e = 0; e < 8; ++e) { xv[e] = x[kc + e]; xx += xv[e] * xv[e]; }
#pragma unroll
        for (int t = 0; t < N_TOPIC; ++t) {
            float d = 0.f;
#pragma unroll
            for (int e = 0; e < 8; ++e) d += xv[e] * tsh[t * N_DIM + kc + e];
            acc[t] += d;
        }
    }
#pragma unroll
    for (int t = 0; t < N_TOPIC; ++t) {
        float Mv = xx + yy[t] - 2.f * acc[t];
        KT[t * N_DOCS + i] = __expf(-3.0f * Mv);       // DT_ALPHA = 3
        MT[t * N_DOCS + i] = Mv;
    }
}

// ---------------- K/M for topic-word (natural [t][j] layout) ----------------

__global__ void __launch_bounds__(256)
fastopic_ktw(const float* __restrict__ word, const float* __restrict__ topic,
             float* __restrict__ K2, float* __restrict__ M2) {
    __shared__ float tsh[N_TOPIC * N_DIM];
    __shared__ float tt[N_TOPIC];
    for (int idx = threadIdx.x; idx < N_TOPIC * N_DIM; idx += 256) tsh[idx] = topic[idx];
    __syncthreads();
    if (threadIdx.x < N_TOPIC) {
        float s = 0.f;
        for (int k = 0; k < N_DIM; ++k) { float t = tsh[threadIdx.x * N_DIM + k]; s += t * t; }
        tt[threadIdx.x] = s;
    }
    __syncthreads();
    const int j = blockIdx.x * 256 + threadIdx.x;
    if (j >= N_VOCAB) return;
    const float* x = word + (size_t)j * N_DIM;
    float acc[N_TOPIC];
#pragma unroll
    for (int t = 0; t < N_TOPIC; ++t) acc[t] = 0.f;
    float ww = 0.f;
    for (int kc = 0; kc < N_DIM; kc += 8) {
        float xv[8];
#pragma unroll
        for (int e = 0; e < 8; ++e) { xv[e] = x[kc + e]; ww += xv[e] * xv[e]; }
#pragma unroll
        for (int t = 0; t < N_TOPIC; ++t) {
            float d = 0.f;
#pragma unroll
            for (int e = 0; e < 8; ++e) d += xv[e] * tsh[t * N_DIM + kc + e];
            acc[t] += d;
        }
    }
#pragma unroll
    for (int t = 0; t < N_TOPIC; ++t) {
        float Mv = tt[t] + ww - 2.f * acc[t];
        K2[(size_t)t * N_VOCAB + j] = __expf(-2.0f * Mv);   // TW_ALPHA = 2
        M2[(size_t)t * N_VOCAB + j] = Mv;
    }
}

// ---------------- doc-topic Sinkhorn (32 blocks x 128, one doc row per thread) ----------------

__global__ void __launch_bounds__(128)
fastopic_sinkhorn_dt(const float* __restrict__ KT, const float* __restrict__ MT,
                     const float* __restrict__ btop, float* __restrict__ acc,
                     unsigned* __restrict__ bars, _Float16* __restrict__ theta_h,
                     float* __restrict__ loss) {
    __shared__ float lred[4 * N_TOPIC];
    __shared__ float bsh[N_TOPIC];
    const int tid  = blockIdx.x * 128 + threadIdx.x;   // doc row, 0..4095
    const int lane = threadIdx.x & 31;
    const int wv   = threadIdx.x >> 5;
    if (threadIdx.x < N_TOPIC) bsh[threadIdx.x] = btop[threadIdx.x];
    __syncthreads();
    float u = 1.0f / (float)N_DOCS;
    float kr[N_TOPIC], v[N_TOPIC];
    for (int it = 0; it < ITERS; ++it) {
#pragma unroll
        for (int t = 0; t < N_TOPIC; ++t) kr[t] = KT[t * N_DOCS + tid];
#pragma unroll
        for (int t = 0; t < N_TOPIC; ++t) {
            float p = wave_sum(kr[t] * u);
            if (lane == 0) lred[wv * N_TOPIC + t] = p;
        }
        __syncthreads();
        if (threadIdx.x < N_TOPIC) {
            float s = lred[threadIdx.x] + lred[N_TOPIC + threadIdx.x]
                    + lred[2 * N_TOPIC + threadIdx.x] + lred[3 * N_TOPIC + threadIdx.x];
            atomicAdd(&acc[it * N_TOPIC + threadIdx.x], s);
        }
        grid_barrier(bars + 0, bars + 1, 32u);
#pragma unroll
        for (int t = 0; t < N_TOPIC; ++t)
            v[t] = bsh[t] / (devload_agent(&acc[it * N_TOPIC + t]) + SEPS);
        float denom = 0.f;
#pragma unroll
        for (int t = 0; t < N_TOPIC; ++t) denom += kr[t] * v[t];
        u = (1.0f / (float)N_DOCS) / (denom + SEPS);
    }
    // u = u^100, v = v^100, kr = K row
    float lsum = 0.f;
#pragma unroll
    for (int t = 0; t < N_TOPIC; ++t) {
        float tr = u * kr[t] * v[t];
        theta_h[(size_t)tid * KPAD + t] = (_Float16)(tr * (float)N_DOCS);
        lsum += tr * MT[t * N_DOCS + tid];
    }
    lsum = wave_sum(lsum);
    if (lane == 0) lred[wv] = lsum;
    __syncthreads();
    if (threadIdx.x == 0) atomicAdd(loss + 0, lred[0] + lred[1] + lred[2] + lred[3]);
}

// ---------------- topic-word Sinkhorn (64 blocks x 256, word columns per thread) ----------------

__global__ void __launch_bounds__(256)
fastopic_sinkhorn_tw(const float* __restrict__ K2, const float* __restrict__ M2,
                     const float* __restrict__ bw, float* __restrict__ acc,
                     unsigned* __restrict__ bars, _Float16* __restrict__ beta_h,
                     float* __restrict__ v2, float* __restrict__ loss) {
    __shared__ float lred[8 * N_TOPIC];
    const int tid  = blockIdx.x * 256 + threadIdx.x;
    const int NTH  = 64 * 256;
    const int lane = threadIdx.x & 31;
    const int wv   = threadIdx.x >> 5;
    float u2[N_TOPIC];
#pragma unroll
    for (int t = 0; t < N_TOPIC; ++t) u2[t] = 1.0f / (float)N_TOPIC;
    for (int it = 0; it < ITERS; ++it) {
        float part[N_TOPIC];
#pragma unroll
        for (int t = 0; t < N_TOPIC; ++t) part[t] = 0.f;
        for (int j = tid; j < N_VOCAB; j += NTH) {
            float denom = 0.f;
#pragma unroll
            for (int t = 0; t < N_TOPIC; ++t) denom += K2[(size_t)t * N_VOCAB + j] * u2[t];
            float vj = bw[j] / (denom + SEPS);
            if (it == ITERS - 1) v2[j] = vj;
#pragma unroll
            for (int t = 0; t < N_TOPIC; ++t) part[t] += K2[(size_t)t * N_VOCAB + j] * vj;
        }
#pragma unroll
        for (int t = 0; t < N_TOPIC; ++t) {
            float p = wave_sum(part[t]);
            if (lane == 0) lred[wv * N_TOPIC + t] = p;
        }
        __syncthreads();
        if (threadIdx.x < N_TOPIC) {
            float s = 0.f;
#pragma unroll
            for (int w2 = 0; w2 < 8; ++w2) s += lred[w2 * N_TOPIC + threadIdx.x];
            atomicAdd(&acc[it * N_TOPIC + threadIdx.x], s);
        }
        grid_barrier(bars + 0, bars + 1, 64u);
#pragma unroll
        for (int t = 0; t < N_TOPIC; ++t)
            u2[t] = (1.0f / (float)N_TOPIC) / (devload_agent(&acc[it * N_TOPIC + t]) + SEPS);
    }
    // u2 = u^100, v2 = v^100 : beta = 50 * u*K*v   (stored [j][KPAD] for WMMA B frags)
    float lsum = 0.f;
    for (int j = tid; j < N_VOCAB; j += NTH) {
        float vj = v2[j];
#pragma unroll
        for (int t = 0; t < N_TOPIC; ++t) {
            float tr = u2[t] * K2[(size_t)t * N_VOCAB + j] * vj;
            beta_h[(size_t)j * KPAD + t] = (_Float16)(tr * (float)N_TOPIC);
            lsum += tr * M2[(size_t)t * N_VOCAB + j];
        }
    }
    lsum = wave_sum(lsum);
    if (lane == 0) lred[wv] = lsum;
    __syncthreads();
    if (threadIdx.x == 0) {
        float s = 0.f;
#pragma unroll
        for (int i = 0; i < 8; ++i) s += lred[i];
        atomicAdd(loss + 1, s);
    }
}

// ---------------- fused WMMA GEMM (theta @ beta) + log + bow dot ----------------
// 16x16 tile per wave, K=64 (50 padded with zeros) -> two v_wmma_f32_16x16x32_f16
// train_bow is streamed once with non-temporal loads (keep theta/beta L2-resident)

__global__ void __launch_bounds__(256)
fastopic_recon_loss(const _Float16* __restrict__ theta_h,
                    const _Float16* __restrict__ beta_h,
                    const float* __restrict__ bow,
                    float* __restrict__ loss) {
    __shared__ float lred[8];
    const int lane = threadIdx.x & 31;
    const int wv   = threadIdx.x >> 5;
    const long wid = (long)blockIdx.x * 8 + wv;        // 0 .. 799999
    const int  mt  = (int)(wid & 255);                 // 256 row tiles (4096/16)
    const long nt  = wid >> 8;                         // 3125 col tiles (50000/16)
    const int  r   = lane & 15;
    const int  g   = lane >> 4;

    // A fragment: row = mt*16 + (lane&15); lane group g selects K = 16*g..16*g+15 (contiguous)
    const _Float16* Ap = theta_h + (size_t)(mt * 16 + r) * KPAD + g * 16;
    v8h alo0 = *(const v8h*)(Ap);          // K = kb..kb+7     (kb = 16*g)
    v8h ahi0 = *(const v8h*)(Ap + 8);      // K = kb+8..kb+15
    v8h alo1 = *(const v8h*)(Ap + 32);     // second k-step (+32)
    v8h ahi1 = *(const v8h*)(Ap + 40);
    v16h a0 = __builtin_shufflevector(alo0, ahi0, 0,1,2,3,4,5,6,7,8,9,10,11,12,13,14,15);
    v16h a1 = __builtin_shufflevector(alo1, ahi1, 0,1,2,3,4,5,6,7,8,9,10,11,12,13,14,15);

    // B fragment: col = nt*16 + (lane&15); halves h -> K = 16*g + h (contiguous in [N][KPAD])
    const _Float16* Bp = beta_h + (size_t)(nt * 16 + r) * KPAD + g * 16;
    v16h b0 = *(const v16h*)(Bp);
    v16h b1 = *(const v16h*)(Bp + 32);

    v8f c = {};
    c = __builtin_amdgcn_wmma_f32_16x16x32_f16(false, a0, false, b0, (short)0, c, false, false);
    c = __builtin_amdgcn_wmma_f32_16x16x32_f16(false, a1, false, b1, (short)0, c, false, false);

    // C layout: element rr -> M = rr + 8*g, N = lane&15
    const size_t jcol = (size_t)nt * 16 + r;
    float accv = 0.f;
#pragma unroll
    for (int rr = 0; rr < 8; ++rr) {
        const size_t irow = (size_t)(mt * 16 + g * 8 + rr);
        float wval = __builtin_nontemporal_load(&bow[irow * N_VOCAB + jcol]);
        accv += wval * __logf(c[rr] + REPS);
    }
    accv = wave_sum(accv);
    if (lane == 0) lred[wv] = accv;
    __syncthreads();
    if (threadIdx.x == 0) {
        float s = 0.f;
#pragma unroll
        for (int i = 0; i < 8; ++i) s += lred[i];
        atomicAdd(loss + 8 + (blockIdx.x & (NSHARD - 1)), s);   // sharded accumulator
    }
}

// ---------------- finalize: out = loss_DT + loss_TW - dsr_sum / N_DOCS ----------------

__global__ void fastopic_finalize(const float* __restrict__ loss, float* __restrict__ out) {
    if (threadIdx.x == 0 && blockIdx.x == 0) {
        float dsr = 0.f;
        for (int i = 0; i < NSHARD; ++i) dsr += loss[8 + i];
        out[0] = loss[0] + loss[1] - dsr * (1.0f / (float)N_DOCS);
    }
}

// ---------------- host launch ----------------

extern "C" void kernel_launch(void* const* d_in, const int* in_sizes, int n_in,
                              void* d_out, int out_size, void* d_ws, size_t ws_size,
                              hipStream_t stream) {
    (void)in_sizes; (void)n_in; (void)out_size; (void)ws_size;
    const float* bow   = (const float*)d_in[0];
    const float* doc   = (const float*)d_in[1];
    const float* word  = (const float*)d_in[2];
    const float* topic = (const float*)d_in[3];
    const float* wwt   = (const float*)d_in[4];
    const float* twt   = (const float*)d_in[5];
    float* out = (float*)d_out;
    char* w = (char*)d_ws;

    size_t off = 0;
    auto take = [&](size_t bytes) { size_t o = off; off = (off + bytes + 255) & ~(size_t)255; return o; };

    // zeroed region (accumulators, pads, loss, barriers)
    const size_t oTheta = take((size_t)N_DOCS * KPAD * sizeof(_Float16));
    const size_t oBeta  = take((size_t)N_VOCAB * KPAD * sizeof(_Float16));
    const size_t oAccDT = take((size_t)ITERS * N_TOPIC * sizeof(float));
    const size_t oAccTW = take((size_t)ITERS * N_TOPIC * sizeof(float));
    const size_t oLoss  = take(256);           // [0]=DT [1]=TW [8..39]=recon shards
    const size_t oBarDT = take(256);
    const size_t oBarTW = take(256);
    const size_t zeroEnd = off;
    // non-zeroed scratch
    const size_t oKTDT = take((size_t)N_TOPIC * N_DOCS * sizeof(float));
    const size_t oMTDT = take((size_t)N_TOPIC * N_DOCS * sizeof(float));
    const size_t oKTW  = take((size_t)N_TOPIC * N_VOCAB * sizeof(float));
    const size_t oMTW  = take((size_t)N_TOPIC * N_VOCAB * sizeof(float));
    const size_t oBW   = take((size_t)N_VOCAB * sizeof(float));
    const size_t oBTOP = take((size_t)N_TOPIC * sizeof(float));
    const size_t oV2   = take((size_t)N_VOCAB * sizeof(float));

    fastopic_zero<<<512, 256, 0, stream>>>((unsigned*)w, zeroEnd / 4);
    fastopic_softmax<<<1, 1024, 0, stream>>>(wwt, twt, (float*)(w + oBW), (float*)(w + oBTOP));
    fastopic_kdt<<<N_DOCS / 256, 256, 0, stream>>>(doc, topic, (float*)(w + oKTDT), (float*)(w + oMTDT));
    fastopic_ktw<<<(N_VOCAB + 255) / 256, 256, 0, stream>>>(word, topic, (float*)(w + oKTW), (float*)(w + oMTW));
    fastopic_sinkhorn_dt<<<32, 128, 0, stream>>>((const float*)(w + oKTDT), (const float*)(w + oMTDT),
                                                 (const float*)(w + oBTOP), (float*)(w + oAccDT),
                                                 (unsigned*)(w + oBarDT), (_Float16*)(w + oTheta),
                                                 (float*)(w + oLoss));
    fastopic_sinkhorn_tw<<<64, 256, 0, stream>>>((const float*)(w + oKTW), (const float*)(w + oMTW),
                                                 (const float*)(w + oBW), (float*)(w + oAccTW),
                                                 (unsigned*)(w + oBarTW), (_Float16*)(w + oBeta),
                                                 (float*)(w + oV2), (float*)(w + oLoss));
    // 256 * 3125 tiles = 800000 waves = 100000 blocks of 8 waves
    fastopic_recon_loss<<<100000, 256, 0, stream>>>((const _Float16*)(w + oTheta),
                                                    (const _Float16*)(w + oBeta),
                                                    bow, (float*)(w + oLoss));
    fastopic_finalize<<<1, 1, 0, stream>>>((const float*)(w + oLoss), out);
}
